// LinearAttention_13932873908469
// MI455X (gfx1250) — compile-verified, matrix operand order checked by reading, not run on hardware
//
#include <hip/hip_runtime.h>
#include <hip/hip_bf16.h>

// Linear attention for MI455X (gfx1250), bf16 WMMA pipeline with
// double-buffered async-LDS staging.
// B=2, L=4096, HIDDEN=4096, H=32, Hkv=8, D=128.

#define HIDDEN   4096
#define SEQ      4096
#define BATCH    2
#define NHEADS   32
#define NKVHEADS 8
#define HDIM     128
#define KVWIDTH  (NKVHEADS * HDIM)   // 1024

typedef __bf16 bf16;
typedef __attribute__((ext_vector_type(8)))  bf16  v8bf;
typedef __attribute__((ext_vector_type(16))) bf16  v16bf;
typedef __attribute__((ext_vector_type(8)))  float v8f;

#define LDS_STRIDE 40                        // 32+8 halves pad: 16B aligned, conflict-breaking
#define TILE_ELEMS (128 * LDS_STRIDE)
#define TILE_BYTES (TILE_ELEMS * 2)          // 10240 B per staged tile

// Fragment per ISA 16-bit A/B 16x32 layout:
// lanes 0-15: K=[0..7],[16..23]; lanes 16-31: K=[8..15],[24..31].
__device__ __forceinline__ v16bf frag_from_lds(const bf16* s, int row, int kb) {
  const v8bf lo = *(const v8bf*)(s + row * LDS_STRIDE + kb);
  const v8bf hi = *(const v8bf*)(s + row * LDS_STRIDE + 16 + kb);
  return __builtin_shufflevector(lo, hi, 0,1,2,3,4,5,6,7,8,9,10,11,12,13,14,15);
}

__device__ __forceinline__ v8f wmma_bf16(v16bf a, v16bf b, v8f c) {
  return __builtin_amdgcn_wmma_f32_16x16x32_bf16(false, a, false, b, (short)0, c, false, false);
}

__device__ __forceinline__ void wait_async_lds() {
  asm volatile("s_wait_asynccnt 0x0" ::: "memory");
}
// Scheduling fence: keep all fragment ds_loads issued before the WMMA chain
// (one dscnt wait instead of one per WMMA).
__device__ __forceinline__ void sched_fence() {
  asm volatile("" ::: "memory");
}

// ---- staging helpers for a 128x32 tile (row stride ld) ----
// f32 source, split into issue (global->VGPR) and commit (cvt+ds_store_b128)
__device__ __forceinline__ void stage_issue_f32(const float* g, long ld, int tid, float4* r) {
#pragma unroll
  for (int i = 0; i < 2; ++i) {
    int c = tid + i * 256;                  // 512 chunks of 8 floats
    int row = c >> 2, col = (c & 3) * 8;
    const float* src = g + (long)row * ld + col;
    r[2*i]   = *(const float4*)src;
    r[2*i+1] = *(const float4*)(src + 4);
  }
}
__device__ __forceinline__ void stage_commit_f32(bf16* s, int tid, const float4* r) {
#pragma unroll
  for (int i = 0; i < 2; ++i) {
    int c = tid + i * 256;
    int row = c >> 2, col = (c & 3) * 8;
    float4 f0 = r[2*i], f1 = r[2*i+1];
    v8bf h = { (bf16)f0.x, (bf16)f0.y, (bf16)f0.z, (bf16)f0.w,
               (bf16)f1.x, (bf16)f1.y, (bf16)f1.z, (bf16)f1.w };
    *(v8bf*)(s + row * LDS_STRIDE + col) = h;
  }
}
// bf16 source: CDNA5 async copy global->LDS, destination = LDS byte offset.
__device__ __forceinline__ void stage_issue_async_bf16(const bf16* g, long ld,
                                                       unsigned lds_byte, int tid) {
#pragma unroll
  for (int i = 0; i < 2; ++i) {
    int c = tid + i * 256;                  // 512 chunks of 16B
    int row = c >> 2, col = (c & 3) * 8;
    const bf16* src = g + (long)row * ld + col;
    unsigned dst = lds_byte + (unsigned)(row * LDS_STRIDE + col) * 2u;
    asm volatile("global_load_async_to_lds_b128 %0, %1, off"
                 :: "v"(dst), "v"(src) : "memory");
  }
}

// Generic C = epilogue(A * B^T).  A: [M x K] row stride lda.  B: [N x K] row-major
// (nn.Linear weight layout).  Block tile 128x128, 8 waves as 2(M) x 4(N),
// wave tile 64x32, K-step 32, double-buffered LDS pipeline.
template<typename TA, typename TB>
__global__ __launch_bounds__(256)
void gemm_wmma_kernel(const void* Av, long lda, long sAb, long sAh,
                      const void* Bv, long sBb, long sBh, int hdivB,
                      const float* bias,
                      const float* colScale, long sSb, long sSh, int hdivS,
                      float* Cf, bf16* Cb, long ldc, long sCb, long sCh,
                      int M, int N, int K, int zdiv, int applyPhi)
{
  const int tid  = threadIdx.x;
  const int lane = tid & 31;
  const int wid  = tid >> 5;
  const int wm   = wid & 1;
  const int wn   = wid >> 1;
  const int m0   = blockIdx.y * 128;
  const int n0   = blockIdx.x * 128;

  const int  z  = blockIdx.z;
  const int  bz = z / zdiv;
  const int  hz = z % zdiv;
  const long aoff = (long)bz * sAb + (long)hz * sAh;
  const long boff = (long)bz * sBb + (long)(hz / hdivB) * sBh;
  const long coff = (long)bz * sCb + (long)hz * sCh;
  const long soff = (long)bz * sSb + (long)(hz / hdivS) * sSh;

  const TA* A = (const TA*)Av + aoff + (long)m0 * lda;
  const TB* B = (const TB*)Bv + boff + (long)n0 * (long)K;

  // Dynamic LDS only (no static) so async-LDS byte offsets are exact.
  // No pointer arrays from shared addresses (ld.lld can't relocate them):
  // ping-pong buffers are computed with integer offsets at point of use.
  extern __shared__ __align__(16) char smem_raw[];
  bf16* smem = (bf16*)smem_raw;   // layout: [A0 | B0 | A1 | B1], each TILE_ELEMS
  const unsigned lds_base = __builtin_amdgcn_groupstaticsize();  // dynamic-LDS start

  v8f acc[4][2];
#pragma unroll
  for (int i = 0; i < 4; ++i)
#pragma unroll
    for (int j = 0; j < 2; ++j)
#pragma unroll
      for (int e = 0; e < 8; ++e) acc[i][j][e] = 0.0f;

  const int r  = lane & 15;
  const int kb = (lane >> 4) * 8;
  constexpr bool asyncA = (sizeof(TA) == 2);
  constexpr bool asyncB = (sizeof(TB) == 2);
  constexpr bool anyAsync = asyncA || asyncB;

  // ---- prologue: stage slab 0 into buffer 0 ----
  if constexpr (asyncA) {
    stage_issue_async_bf16((const bf16*)A, lda, lds_base, tid);
  } else {
    float4 ra[4];
    stage_issue_f32((const float*)A, lda, tid, ra);
    stage_commit_f32(smem, tid, ra);
  }
  if constexpr (asyncB) {
    stage_issue_async_bf16((const bf16*)B, (long)K, lds_base + TILE_BYTES, tid);
  } else {
    float4 rb[4];
    stage_issue_f32((const float*)B, (long)K, tid, rb);
    stage_commit_f32(smem + TILE_ELEMS, tid, rb);
  }
  if (anyAsync) wait_async_lds();
  __syncthreads();

  const int nk = K >> 5;
  for (int it = 0; it < nk; ++it) {
    const int  k0  = it << 5;
    const int  cur = it & 1;
    const int  nxt = cur ^ 1;
    const bool hasNext = (it + 1 < nk);
    const unsigned ldsA_nxt = lds_base + (unsigned)nxt * 2u * TILE_BYTES;
    const unsigned ldsB_nxt = ldsA_nxt + TILE_BYTES;

    // Issue next slab early: async copies run under the WMMA chain.
    float4 ra[4], rb[4];
    if (hasNext) {
      if constexpr (asyncA) stage_issue_async_bf16((const bf16*)A + k0 + 32, lda, ldsA_nxt, tid);
      else                  stage_issue_f32((const float*)A + k0 + 32, lda, tid, ra);
      if constexpr (asyncB) stage_issue_async_bf16((const bf16*)B + k0 + 32, (long)K, ldsB_nxt, tid);
      else                  stage_issue_f32((const float*)B + k0 + 32, (long)K, tid, rb);
    }
    if (k0 + 64 < K) {  // global_prefetch_b8 two slabs ahead
      __builtin_prefetch((const void*)(A + (long)(tid >> 1) * lda + k0 + 64), 0, 0);
      __builtin_prefetch((const void*)(B + (long)(tid >> 1) * (long)K + k0 + 64), 0, 0);
    }

    // Load all fragments, fence, then 8 back-to-back WMMAs.
    const bf16* As = smem + cur * 2 * TILE_ELEMS;
    const bf16* Bs = As + TILE_ELEMS;
    v16bf af[4], bfg[2];
#pragma unroll
    for (int mt = 0; mt < 4; ++mt)
      af[mt] = frag_from_lds(As, wm * 64 + mt * 16 + r, kb);
#pragma unroll
    for (int nt = 0; nt < 2; ++nt)
      bfg[nt] = frag_from_lds(Bs, wn * 32 + nt * 16 + r, kb);
    sched_fence();
#pragma unroll
    for (int nt = 0; nt < 2; ++nt)
#pragma unroll
      for (int mt = 0; mt < 4; ++mt)
        acc[mt][nt] = wmma_bf16(af[mt], bfg[nt], acc[mt][nt]);

    if (hasNext) {
      bf16* AsN = smem + nxt * 2 * TILE_ELEMS;
      if constexpr (!asyncA) stage_commit_f32(AsN, tid, ra);
      if constexpr (!asyncB) stage_commit_f32(AsN + TILE_ELEMS, tid, rb);
      if (anyAsync) wait_async_lds();     // async slab must land before publish
      __syncthreads();
    }
  }

  // Epilogue.  C/D layout: lane l -> N = l&15, M = vgpr + 8*(l>>4)
  const int hsel = lane >> 4;
#pragma unroll
  for (int mt = 0; mt < 4; ++mt) {
#pragma unroll
    for (int nt = 0; nt < 2; ++nt) {
#pragma unroll
      for (int v = 0; v < 8; ++v) {
        const int  mloc = wm * 64 + mt * 16 + hsel * 8 + v;
        const int  nloc = wn * 32 + nt * 16 + r;
        const long m = m0 + mloc;
        const long n = n0 + nloc;
        float val = acc[mt][nt][v];
        if (bias)     val += bias[n];
        if (applyPhi) val = (val > 0.0f) ? (val + 1.0f) : __expf(val);  // elu(x)+1
        if (colScale) val *= colScale[soff + n];
        if (Cf) Cf[coff + m * ldc + n] = val;
        else    Cb[coff + m * ldc + n] = (bf16)val;
      }
    }
  }
}

// KV[b,kvh][v][d] = sum_l V[b,l,kvh,v] * Kphi[b,l,kvh,d].  One workgroup per (b,kvh).
// M=N=128, K=L=4096; operands staged transposed so fragment reads are b128 ds loads.
__global__ __launch_bounds__(256)
void kv_summary_kernel(const bf16* __restrict__ Kphi, const bf16* __restrict__ Vb,
                       bf16* __restrict__ KVb)
{
  const int z    = blockIdx.x;         // 0..15
  const int b    = z >> 3;
  const int kvh  = z & 7;
  const int tid  = threadIdx.x;
  const int lane = tid & 31;
  const int w    = tid >> 5;           // wave -> v-tile row

  __shared__ alignas(16) bf16 Vt[128 * LDS_STRIDE];  // [v][l]
  __shared__ alignas(16) bf16 Kt[128 * LDS_STRIDE];  // [d][l]

  const bf16* Kg = Kphi + (long)b * SEQ * KVWIDTH + kvh * HDIM;
  const bf16* Vg = Vb   + (long)b * SEQ * KVWIDTH + kvh * HDIM;

  v8f acc[8];
#pragma unroll
  for (int i = 0; i < 8; ++i)
#pragma unroll
    for (int e = 0; e < 8; ++e) acc[i][e] = 0.0f;

  const int r  = lane & 15;
  const int kb = (lane >> 4) * 8;

  for (int l0 = 0; l0 < SEQ; l0 += 32) {
#pragma unroll
    for (int i = 0; i < 16; ++i) {
      int e = tid + i * 256;           // 4096 = 32 l-rows x 128 cols
      int l = e >> 7;
      int c = e & 127;
      Vt[c * LDS_STRIDE + l] = Vg[(long)(l0 + l) * KVWIDTH + c];
      Kt[c * LDS_STRIDE + l] = Kg[(long)(l0 + l) * KVWIDTH + c];
    }
    __syncthreads();
    v16bf a = frag_from_lds(Vt, w * 16 + r, kb);
    v16bf bb[8];
#pragma unroll
    for (int nt = 0; nt < 8; ++nt)
      bb[nt] = frag_from_lds(Kt, nt * 16 + r, kb);
    sched_fence();
#pragma unroll
    for (int nt = 0; nt < 8; ++nt)
      acc[nt] = wmma_bf16(a, bb[nt], acc[nt]);
    __syncthreads();
  }

  const int hsel = lane >> 4;
#pragma unroll
  for (int nt = 0; nt < 8; ++nt)
#pragma unroll
    for (int v = 0; v < 8; ++v) {
      int vr = w * 16 + hsel * 8 + v;
      int d  = nt * 16 + r;
      KVb[((long)z * HDIM + vr) * HDIM + d] = (bf16)acc[nt][v];
    }
}

// scale[b, c] = 1 / (sum_l Kphi[b, l, c] + 1e-10)
__global__ __launch_bounds__(256)
void norm_scale_kernel(const bf16* __restrict__ Kphi, float* __restrict__ scale)
{
  int g = blockIdx.x * blockDim.x + threadIdx.x;   // 2048 threads
  int b = g >> 10;
  int c = g & 1023;
  const bf16* p = Kphi + (long)b * SEQ * KVWIDTH + c;
  float s = 0.0f;
  for (int l = 0; l < SEQ; ++l) s += (float)p[(long)l * KVWIDTH];
  scale[g] = 1.0f / (s + 1e-10f);
}

extern "C" void kernel_launch(void* const* d_in, const int* in_sizes, int n_in,
                              void* d_out, int out_size, void* d_ws, size_t ws_size,
                              hipStream_t stream)
{
  const float* X  = (const float*)d_in[0];
  const float* Wq = (const float*)d_in[1];
  const float* bq = (const float*)d_in[2];
  const float* Wk = (const float*)d_in[3];
  const float* bk = (const float*)d_in[4];
  const float* Wv = (const float*)d_in[5];
  const float* bv = (const float*)d_in[6];
  const float* Wo = (const float*)d_in[7];
  const float* bo = (const float*)d_in[8];
  float* Out = (float*)d_out;

  char* w = (char*)d_ws;
  bf16*  Qphi = (bf16*)w;  w += (size_t)BATCH * SEQ * HIDDEN  * 2;   // 64 MB
  bf16*  Kphi = (bf16*)w;  w += (size_t)BATCH * SEQ * KVWIDTH * 2;   // 16 MB
  bf16*  Vbuf = (bf16*)w;  w += (size_t)BATCH * SEQ * KVWIDTH * 2;   // 16 MB
  bf16*  KVb  = (bf16*)w;  w += (size_t)BATCH * NKVHEADS * HDIM * HDIM * 2;  // 0.5 MB
  float* scl  = (float*)w; w += (size_t)BATCH * KVWIDTH * 4;         // 8 KB
  bf16*  Attn = (bf16*)w;                                            // 64 MB

  const long BL = (long)BATCH * SEQ;        // 8192
  const size_t shmem = 4 * TILE_BYTES;      // 40960 B: double-buffered A+B tiles

  // 1) Qphi = phi(X @ Wq^T + bq) -> bf16
  gemm_wmma_kernel<float, float><<<dim3(HIDDEN / 128, BL / 128, 1), 256, shmem, stream>>>(
      X, HIDDEN, 0, 0,  Wq, 0, 0, 1,  bq,  nullptr, 0, 0, 1,
      nullptr, Qphi, HIDDEN, 0, 0,  (int)BL, HIDDEN, HIDDEN, 1, 1);

  // 2) Kphi = phi(X @ Wk^T + bk) -> bf16
  gemm_wmma_kernel<float, float><<<dim3(KVWIDTH / 128, BL / 128, 1), 256, shmem, stream>>>(
      X, HIDDEN, 0, 0,  Wk, 0, 0, 1,  bk,  nullptr, 0, 0, 1,
      nullptr, Kphi, KVWIDTH, 0, 0,  (int)BL, KVWIDTH, HIDDEN, 1, 1);

  // 3) V = X @ Wv^T + bv -> bf16
  gemm_wmma_kernel<float, float><<<dim3(KVWIDTH / 128, BL / 128, 1), 256, shmem, stream>>>(
      X, HIDDEN, 0, 0,  Wv, 0, 0, 1,  bv,  nullptr, 0, 0, 1,
      nullptr, Vbuf, KVWIDTH, 0, 0,  (int)BL, KVWIDTH, HIDDEN, 1, 0);

  // 4) per-channel normalizer reciprocal
  norm_scale_kernel<<<dim3(8), 256, 0, stream>>>(Kphi, scl);

  // 5) KV summary per (b, kv-head)
  kv_summary_kernel<<<dim3(BATCH * NKVHEADS), 256, 0, stream>>>(Kphi, Vbuf, KVb);

  // 6) attn[b,h] = (Qphi[b,h] @ KV[b,h/4]^T) * scale[b,h/4] -> bf16 (all-async staging)
  gemm_wmma_kernel<bf16, bf16><<<dim3(1, SEQ / 128, BATCH * NHEADS), 256, shmem, stream>>>(
      Qphi, HIDDEN, (long)SEQ * HIDDEN, HDIM,
      KVb, (long)NKVHEADS * HDIM * HDIM, (long)HDIM * HDIM, 4,
      nullptr,
      scl, KVWIDTH, HDIM, 4,
      nullptr, Attn, HIDDEN, (long)SEQ * HIDDEN, HDIM,
      SEQ, HDIM, HDIM, NHEADS, 0);

  // 7) Out = Attn @ Wo^T + bo -> f32 (A async-LDS, B f32 convert-in-VGPR)
  gemm_wmma_kernel<bf16, float><<<dim3(HIDDEN / 128, BL / 128, 1), 256, shmem, stream>>>(
      Attn, HIDDEN, 0, 0,  Wo, 0, 0, 1,  bo,  nullptr, 0, 0, 1,
      Out, nullptr, HIDDEN, 0, 0,  (int)BL, HIDDEN, HIDDEN, 1, 0);
}
